// RESMAv2Fast_40269613367379
// MI455X (gfx1250) — matrix operand order repaired and checked by reading, not compiled, not run on hardware
//
#include <hip/hip_runtime.h>
#include <hip/hip_bf16.h>

typedef __attribute__((ext_vector_type(16))) _Float16 v16h;
typedef __attribute__((ext_vector_type(8)))  _Float16 v8h;
typedef __attribute__((ext_vector_type(8)))  float    v8f;

enum : int {
    NN   = 100000,
    EE   = 1600000,
    HID  = 128,
    HEADS = 4,
    HDIM = 32,
};
#define NEG_SLOPE 0.2f
#define LN_EPS 1e-5f

// ---------- helpers ----------
__device__ __forceinline__ unsigned f2ord(float f) {
    unsigned u = __float_as_uint(f);
    return (u & 0x80000000u) ? ~u : (u | 0x80000000u);
}
__device__ __forceinline__ float ord2f(unsigned u) {
    return (u & 0x80000000u) ? __uint_as_float(u & 0x7FFFFFFFu) : __uint_as_float(~u);
}
__device__ __forceinline__ void edge_sd(const int* __restrict__ ei, int item, int& s, int& d) {
    if (item < EE) { s = ei[item]; d = ei[EE + item]; }
    else           { s = item - EE; d = item - EE; }     // self loops
}

// ---------- f32 -> f16 elementwise ----------
__global__ void cvt_f16_kernel(const float* __restrict__ src, _Float16* __restrict__ dst, int n) {
    int t = blockIdx.x * blockDim.x + threadIdx.x;
    if (t < n) dst[t] = (_Float16)src[t];
}

// ---------- pack W[M x K] (row-major f32) into WMMA B-fragment layout (f16) ----------
// Bp[((ct*(K/32)+kk)*32 + lane)*16 + j] = W[(ct*16 + lane%16)*K + kk*32 + (j/8)*16 + (lane/16)*8 + j%8]
__global__ void pack_b_kernel(const float* __restrict__ W, _Float16* __restrict__ Bp, int M, int K) {
    int t = blockIdx.x * blockDim.x + threadIdx.x;
    if (t >= M * K) return;
    int j    = t & 15;
    int lane = (t >> 4) & 31;
    int rest = t >> 9;
    int ksteps = K >> 5;
    int kk = rest % ksteps;
    int ct = rest / ksteps;
    int n = (ct << 4) + (lane & 15);
    int k = (kk << 5) + ((j >> 3) << 4) + ((lane >> 4) << 3) + (j & 7);
    Bp[t] = (_Float16)W[(size_t)n * K + k];
}

// ---------- WMMA GEMM: C[nRows x M] = A[nRows x K] * Bp + bias ----------
__global__ void gemm_wmma_kernel(const _Float16* __restrict__ A,
                                 const _Float16* __restrict__ Bp,
                                 const float* __restrict__ bias,
                                 float* __restrict__ C, int ldc,
                                 _Float16* __restrict__ C16, int ldc16, int c16off,
                                 int nRows, int K, int M, int relu)
{
    const int tilesM = M >> 4;
    const int totalTiles = (nRows >> 4) * tilesM;
    int wave = blockIdx.x * (blockDim.x >> 5) + (threadIdx.x >> 5);
    if (wave >= totalTiles) return;
    int lane = threadIdx.x & 31;
    int rowTile = wave / tilesM;
    int colTile = wave % tilesM;
    int row = (rowTile << 4) + (lane & 15);
    int hs  = lane >> 4;
    int kSteps = K >> 5;
    const _Float16* Arow = A + (size_t)row * K;
    const _Float16* Bt   = Bp + ((size_t)colTile * kSteps * 32 + lane) * 16;
    v8f acc = {};
    for (int kk = 0; kk < kSteps; ++kk) {
        int k0 = kk << 5;
        __builtin_prefetch(Arow + k0 + 64, 0, 1);
        v8h a0 = *(const v8h*)(Arow + k0 + hs * 8);
        v8h a1 = *(const v8h*)(Arow + k0 + 16 + hs * 8);
        v16h a = __builtin_shufflevector(a0, a1, 0,1,2,3,4,5,6,7,8,9,10,11,12,13,14,15);
        v16h b = *(const v16h*)(Bt + (size_t)kk * 32 * 16);
        acc = __builtin_amdgcn_wmma_f32_16x16x32_f16(false, a, false, b, (short)0, acc,
                                                     false, false);
    }
    int n = (colTile << 4) + (lane & 15);
    float bv = bias ? bias[n] : 0.0f;
    int mbase = (rowTile << 4) + (hs << 3);
    #pragma unroll
    for (int r2 = 0; r2 < 8; ++r2) {
        float v = acc[r2] + bv;
        if (relu) v = v > 0.0f ? v : 0.0f;
        int m = mbase + r2;
        if (C)   C  [(size_t)m * ldc   + n]          = v;
        if (C16) C16[(size_t)m * ldc16 + c16off + n] = (_Float16)v;
    }
}

// ---------- degree & dinv ----------
__global__ void deg_kernel(const int* __restrict__ ei, float* __restrict__ deg) {
    int t = blockIdx.x * blockDim.x + threadIdx.x;
    if (t < EE) atomicAdd(&deg[ei[EE + t]], 1.0f);
}
__global__ void dinv_kernel(const float* __restrict__ deg, float* __restrict__ dinv) {
    int t = blockIdx.x * blockDim.x + threadIdx.x;
    if (t < NN) { float d = deg[t]; dinv[t] = d > 0.0f ? rsqrtf(d) : 0.0f; }
}

// ---------- E8 scatter: agg[row] += h[col] * dinv[row]*dinv[col], wave per edge ----------
__global__ void e8_scatter_kernel(const int* __restrict__ ei, const float* __restrict__ dinv,
                                  const float* __restrict__ h, float* __restrict__ agg)
{
    int wid = blockIdx.x * (blockDim.x >> 5) + (threadIdx.x >> 5);
    if (wid >= EE) return;
    int lane = threadIdx.x & 31;
    int r = ei[wid], c = ei[EE + wid];
    float val = dinv[r] * dinv[c];
    int d0 = lane * 4;
    const float* hp = h   + (size_t)c * HID + d0;
    float*       ap = agg + (size_t)r * HID + d0;
    #pragma unroll
    for (int t = 0; t < 4; ++t) atomicAdd(ap + t, hp[t] * val);
}

// ---------- attention per-node scores ----------
__global__ void attn_node_kernel(const float* __restrict__ g,
                                 const float* __restrict__ att_src,
                                 const float* __restrict__ att_dst,
                                 float* __restrict__ asrc, float* __restrict__ adst)
{
    int t = blockIdx.x * blockDim.x + threadIdx.x;
    if (t >= NN * HEADS) return;
    int n = t >> 2, hh = t & 3;
    const float* gp = g + (size_t)n * HID + hh * HDIM;
    float ss = 0.0f, sd = 0.0f;
    #pragma unroll
    for (int c = 0; c < HDIM; ++c) {
        float gv = gp[c];
        ss += gv * att_src[hh * HDIM + c];
        sd += gv * att_dst[hh * HDIM + c];
    }
    asrc[t] = ss; adst[t] = sd;
}

__global__ void minit_kernel(unsigned* __restrict__ m) {
    int t = blockIdx.x * blockDim.x + threadIdx.x;
    if (t < NN * HEADS) m[t] = 0x007FFFFFu;   // f2ord(-inf)
}

// ---------- softmax pass1: leaky_relu + scatter max ----------
__global__ void attn_pass1_kernel(const int* __restrict__ ei,
                                  const float* __restrict__ asrc, const float* __restrict__ adst,
                                  float* __restrict__ elin, unsigned* __restrict__ m)
{
    int t = blockIdx.x * blockDim.x + threadIdx.x;
    if (t >= (EE + NN) * HEADS) return;
    int item = t >> 2, hh = t & 3;
    int s, d; edge_sd(ei, item, s, d);
    float e = asrc[s * HEADS + hh] + adst[d * HEADS + hh];
    e = e > 0.0f ? e : NEG_SLOPE * e;
    elin[t] = e;
    atomicMax(&m[d * HEADS + hh], f2ord(e));
}

// ---------- softmax pass2: exp + scatter sum ----------
__global__ void attn_pass2_kernel(const int* __restrict__ ei,
                                  float* __restrict__ elin,
                                  const unsigned* __restrict__ m, float* __restrict__ s)
{
    int t = blockIdx.x * blockDim.x + threadIdx.x;
    if (t >= (EE + NN) * HEADS) return;
    int item = t >> 2, hh = t & 3;
    int sIdx, d; edge_sd(ei, item, sIdx, d);
    float w = __expf(elin[t] - ord2f(m[d * HEADS + hh]));
    elin[t] = w;
    atomicAdd(&s[d * HEADS + hh], w);
}

// ---------- softmax pass3: scatter messages, wave per (edge+self-loop) ----------
__global__ void attn_pass3_kernel(const int* __restrict__ ei,
                                  const float* __restrict__ elin, const float* __restrict__ ssum,
                                  const float* __restrict__ g, float* __restrict__ xgat)
{
    int wid = blockIdx.x * (blockDim.x >> 5) + (threadIdx.x >> 5);
    if (wid >= EE + NN) return;
    int lane = threadIdx.x & 31;
    int s, d; edge_sd(ei, wid, s, d);
    int hh = lane >> 3;
    int c0 = (lane & 7) * 4;
    float alpha = elin[wid * HEADS + hh] / (ssum[d * HEADS + hh] + 1e-16f);
    const float* gp = g    + (size_t)s * HID + hh * HDIM + c0;
    float*       xp = xgat + (size_t)d * HID + hh * HDIM + c0;
    #pragma unroll
    for (int t = 0; t < 4; ++t) atomicAdd(xp + t, gp[t] * alpha);
}

// ---------- xgat + b_gat -> f16 into concat[:,128:256] ----------
__global__ void gat_pack_kernel(const float* __restrict__ xgat, const float* __restrict__ b_gat,
                                _Float16* __restrict__ concat16)
{
    int t = blockIdx.x * blockDim.x + threadIdx.x;
    if (t >= NN * HID) return;
    int n = t >> 7, d = t & 127;
    concat16[(size_t)n * (2 * HID) + HID + d] = (_Float16)(xgat[t] + b_gat[d]);
}

// ---------- residual + LayerNorm + ReLU (wave per node), writes f16 ----------
__global__ void ln_kernel(const float* __restrict__ f, const float* __restrict__ h,
                          const float* __restrict__ gamma, const float* __restrict__ beta,
                          _Float16* __restrict__ z16)
{
    int node = blockIdx.x * (blockDim.x >> 5) + (threadIdx.x >> 5);
    if (node >= NN) return;
    int lane = threadIdx.x & 31;
    float v[4];
    float s = 0.0f;
    #pragma unroll
    for (int t = 0; t < 4; ++t) {
        int d = lane * 4 + t;
        v[t] = f[(size_t)node * HID + d] + h[(size_t)node * HID + d];
        s += v[t];
    }
    #pragma unroll
    for (int o = 16; o >= 1; o >>= 1) s += __shfl_xor(s, o, 32);
    float mu = s * (1.0f / HID);
    float q = 0.0f;
    #pragma unroll
    for (int t = 0; t < 4; ++t) { float dd = v[t] - mu; q += dd * dd; }
    #pragma unroll
    for (int o = 16; o >= 1; o >>= 1) q += __shfl_xor(q, o, 32);
    float rstd = rsqrtf(q * (1.0f / HID) + LN_EPS);
    #pragma unroll
    for (int t = 0; t < 4; ++t) {
        int d = lane * 4 + t;
        float z = (v[t] - mu) * rstd * gamma[d] + beta[d];
        z = z > 0.0f ? z : 0.0f;
        z16[(size_t)node * HID + d] = (_Float16)z;
    }
}

// ---------- final readout: sigmoid(r @ W_r2^T + b) ----------
__global__ void readout_kernel(const float* __restrict__ r, const float* __restrict__ w2,
                               const float* __restrict__ b2, float* __restrict__ out)
{
    int t = blockIdx.x * blockDim.x + threadIdx.x;
    if (t >= NN) return;
    float acc = b2[0];
    const float* rp = r + (size_t)t * 64;
    #pragma unroll 8
    for (int j = 0; j < 64; ++j) acc += rp[j] * w2[j];
    out[t] = 1.0f / (1.0f + __expf(-acc));
}

// ============================================================================
extern "C" void kernel_launch(void* const* d_in, const int* in_sizes, int n_in,
                              void* d_out, int out_size, void* d_ws, size_t ws_size,
                              hipStream_t stream)
{
    const float* x       = (const float*)d_in[0];
    const int*   ei      = (const int*)  d_in[1];
    const float* W_emb   = (const float*)d_in[2];
    const float* b_emb   = (const float*)d_in[3];
    const float* W_e8    = (const float*)d_in[4];
    const float* W_gat   = (const float*)d_in[5];
    const float* att_src = (const float*)d_in[6];
    const float* att_dst = (const float*)d_in[7];
    const float* b_gat   = (const float*)d_in[8];
    const float* W_fus   = (const float*)d_in[9];
    const float* b_fus   = (const float*)d_in[10];
    const float* gamma   = (const float*)d_in[11];
    const float* beta    = (const float*)d_in[12];
    const float* W_r1    = (const float*)d_in[13];
    const float* b_r1    = (const float*)d_in[14];
    const float* W_r2    = (const float*)d_in[15];
    const float* b_r2    = (const float*)d_in[16];
    float* out = (float*)d_out;

    // ---- carve workspace ----
    char* w = (char*)d_ws;
    size_t off = 0;
    auto take = [&](size_t bytes) -> char* {
        char* p = w + off;
        off += (bytes + 255) & ~(size_t)255;
        return p;
    };
    float*    deg      = (float*)   take((size_t)NN * 4);
    float*    dinv     = (float*)   take((size_t)NN * 4);
    _Float16* x16      = (_Float16*)take((size_t)NN * HID * 2);   // reused as z16 later
    float*    h        = (float*)   take((size_t)NN * HID * 4);
    _Float16* h16      = (_Float16*)take((size_t)NN * HID * 2);
    float*    bufA     = (float*)   take((size_t)NN * HID * 4);   // agg, then xgat
    _Float16* buf16    = (_Float16*)take((size_t)NN * HID * 2);   // agg16, then r (f32, N*64)
    _Float16* concat16 = (_Float16*)take((size_t)NN * 2 * HID * 2);
    float*    bufG     = (float*)   take((size_t)NN * HID * 4);   // g, then f
    float*    asrc     = (float*)   take((size_t)NN * HEADS * 4);
    float*    adst     = (float*)   take((size_t)NN * HEADS * 4);
    unsigned* mbuf     = (unsigned*)take((size_t)NN * HEADS * 4);
    float*    sbuf     = (float*)   take((size_t)NN * HEADS * 4);
    float*    elin     = (float*)   take((size_t)(EE + NN) * HEADS * 4);
    _Float16* BpEmb    = (_Float16*)take(128 * 128 * 2);
    _Float16* BpE8     = (_Float16*)take(128 * 128 * 2);
    _Float16* BpGat    = (_Float16*)take(128 * 128 * 2);
    _Float16* BpFus    = (_Float16*)take(128 * 256 * 2);
    _Float16* BpR1     = (_Float16*)take(64 * 128 * 2);

    const int B = 256;
    auto cdiv = [](long long a, long long b) { return (unsigned)((a + b - 1) / b); };

    // ---- weight packing (tiny) ----
    pack_b_kernel<<<cdiv(128 * 128, B), B, 0, stream>>>(W_emb, BpEmb, 128, 128);
    pack_b_kernel<<<cdiv(128 * 128, B), B, 0, stream>>>(W_e8,  BpE8,  128, 128);
    pack_b_kernel<<<cdiv(128 * 128, B), B, 0, stream>>>(W_gat, BpGat, 128, 128);
    pack_b_kernel<<<cdiv(128 * 256, B), B, 0, stream>>>(W_fus, BpFus, 128, 256);
    pack_b_kernel<<<cdiv(64  * 128, B), B, 0, stream>>>(W_r1,  BpR1,  64,  128);

    // ---- degree / dinv ----
    hipMemsetAsync(deg, 0, (size_t)NN * 4, stream);
    deg_kernel <<<cdiv(EE, B), B, 0, stream>>>(ei, deg);
    dinv_kernel<<<cdiv(NN, B), B, 0, stream>>>(deg, dinv);

    // ---- embedding GEMM: h = x @ W_emb^T + b_emb  (f32 + f16 out) ----
    cvt_f16_kernel<<<cdiv((size_t)NN * HID, B), B, 0, stream>>>(x, x16, NN * HID);
    {
        int tiles = (NN / 16) * (128 / 16);
        gemm_wmma_kernel<<<cdiv(tiles, 8), B, 0, stream>>>(
            x16, BpEmb, b_emb, h, HID, h16, HID, 0, NN, 128, 128, 0);
    }

    // ---- E8: scatter-agg then GEMM into concat[:,0:128] ----
    hipMemsetAsync(bufA, 0, (size_t)NN * HID * 4, stream);
    e8_scatter_kernel<<<cdiv(EE, 8), B, 0, stream>>>(ei, dinv, h, bufA);
    cvt_f16_kernel<<<cdiv((size_t)NN * HID, B), B, 0, stream>>>(bufA, buf16, NN * HID);
    {
        int tiles = (NN / 16) * (128 / 16);
        gemm_wmma_kernel<<<cdiv(tiles, 8), B, 0, stream>>>(
            buf16, BpE8, nullptr, nullptr, 0, concat16, 2 * HID, 0, NN, 128, 128, 0);
    }

    // ---- GAT: g = h @ W_gat^T ----
    {
        int tiles = (NN / 16) * (128 / 16);
        gemm_wmma_kernel<<<cdiv(tiles, 8), B, 0, stream>>>(
            h16, BpGat, nullptr, bufG, HID, nullptr, 0, 0, NN, 128, 128, 0);
    }
    attn_node_kernel<<<cdiv((size_t)NN * HEADS, B), B, 0, stream>>>(bufG, att_src, att_dst, asrc, adst);

    // ---- scatter softmax over (E + N self-loop) items ----
    minit_kernel<<<cdiv((size_t)NN * HEADS, B), B, 0, stream>>>(mbuf);
    hipMemsetAsync(sbuf, 0, (size_t)NN * HEADS * 4, stream);
    attn_pass1_kernel<<<cdiv((size_t)(EE + NN) * HEADS, B), B, 0, stream>>>(ei, asrc, adst, elin, mbuf);
    attn_pass2_kernel<<<cdiv((size_t)(EE + NN) * HEADS, B), B, 0, stream>>>(ei, elin, mbuf, sbuf);
    hipMemsetAsync(bufA, 0, (size_t)NN * HID * 4, stream);   // bufA now = xgat
    attn_pass3_kernel<<<cdiv(EE + NN, 8), B, 0, stream>>>(ei, elin, sbuf, bufG, bufA);
    gat_pack_kernel<<<cdiv((size_t)NN * HID, B), B, 0, stream>>>(bufA, b_gat, concat16);

    // ---- fusion GEMM: f = concat @ W_fus^T + b_fus  (bufG now = f) ----
    {
        int tiles = (NN / 16) * (128 / 16);
        gemm_wmma_kernel<<<cdiv(tiles, 8), B, 0, stream>>>(
            concat16, BpFus, b_fus, bufG, HID, nullptr, 0, 0, NN, 256, 128, 0);
    }

    // ---- residual + LN + ReLU -> z16 (reuse x16) ----
    ln_kernel<<<cdiv(NN, 8), B, 0, stream>>>(bufG, h, gamma, beta, x16);

    // ---- readout GEMM: r = relu(z @ W_r1^T + b_r1)  (buf16 reused as f32 [N,64]) ----
    {
        int tiles = (NN / 16) * (64 / 16);
        gemm_wmma_kernel<<<cdiv(tiles, 8), B, 0, stream>>>(
            x16, BpR1, b_r1, (float*)buf16, 64, nullptr, 0, 0, NN, 128, 64, 1);
    }
    readout_kernel<<<cdiv(NN, B), B, 0, stream>>>((const float*)buf16, W_r2, b_r2, out);
}
